// MultiHeadAttentionLayer_25769804214
// MI455X (gfx1250) — compile-verified, matrix operand order checked by reading, not run on hardware
//
#include <hip/hip_runtime.h>
#include <hip/hip_bf16.h>

typedef __attribute__((ext_vector_type(16))) _Float16 v16h;
typedef __attribute__((ext_vector_type(8)))  float    v8f;

#define NB    4
#define LSEQ  2048
#define HDIM  1024
#define NHEAD 16
#define CDIM  64

static __device__ __forceinline__ v8f wmma16(v16h a, v16h b, v8f c) {
  return __builtin_amdgcn_wmma_f32_16x16x32_f16(false, a, false, b, (short)0, c, false, false);
}

// A-style fragment (16-dim index on lane&15, K striped within lane) from
// row-major s[(i0+m)*ld + k0+k].  Also the B-fragment loader for sources
// stored [n][k] row-major (e.g. K^T from K stored [key][c]).
static __device__ __forceinline__ v16h frag_mk(const _Float16* s, int i0, int ld, int k0) {
  int lane = threadIdx.x & 31;
  const _Float16* row = s + (size_t)(i0 + (lane & 15)) * ld + k0 + ((lane >> 4) << 3);
  v16h f;
#pragma unroll
  for (int j = 0; j < 16; ++j)
    f[j] = row[((j >> 3) << 4) + (j & 7)];
  return f;
}

// B-style fragment (32 x 16) from K-major source s[(k0+k)*ld + n0+n].
static __device__ __forceinline__ v16h frag_kn(const _Float16* s, int n0, int ld, int k0) {
  int lane = threadIdx.x & 31;
  const _Float16* col = s + n0 + (lane & 15) + (size_t)(k0 + ((lane >> 4) << 3)) * ld;
  v16h f;
#pragma unroll
  for (int j = 0; j < 16; ++j)
    f[j] = col[(size_t)(((j >> 3) << 4) + (j & 7)) * ld];
  return f;
}

// ---- CDNA5 async global->LDS staging (ASYNCcnt path, no VGPR bounce) ------
// Copies 2 KB per instruction across 128 threads (16 B/lane); INST_OFFSET is
// added to BOTH the LDS and global addresses, so one LDS/VGPR pair covers a
// contiguous tile.  Generic shared pointers carry the LDS byte offset in
// their low 32 bits.
static __device__ __forceinline__ void async_kv_tile(const _Float16* gK, const _Float16* gV,
                                                     _Float16* sK, _Float16* sV, int tid) {
  unsigned lk = (unsigned)(size_t)sK + (unsigned)tid * 16u;
  unsigned lv = (unsigned)(size_t)sV + (unsigned)tid * 16u;
  unsigned go = (unsigned)tid * 16u;
  asm volatile(
      "global_load_async_to_lds_b128 %0, %2, %3\n\t"
      "global_load_async_to_lds_b128 %0, %2, %3 offset:2048\n\t"
      "global_load_async_to_lds_b128 %1, %2, %4\n\t"
      "global_load_async_to_lds_b128 %1, %2, %4 offset:2048"
      :: "v"(lk), "v"(lv), "v"(go), "s"(gK), "s"(gV)
      : "memory");
}

static __device__ __forceinline__ void async_q_tile(const _Float16* gQ, _Float16* sQ, int tid) {
  unsigned lq = (unsigned)(size_t)sQ + (unsigned)tid * 16u;
  unsigned go = (unsigned)tid * 16u;
  asm volatile(
      "global_load_async_to_lds_b128 %0, %1, %2\n\t"
      "global_load_async_to_lds_b128 %0, %1, %2 offset:2048\n\t"
      "global_load_async_to_lds_b128 %0, %1, %2 offset:4096\n\t"
      "global_load_async_to_lds_b128 %0, %1, %2 offset:6144"
      :: "v"(lq), "v"(go), "s"(gQ)
      : "memory");
}

// ---------------------------------------------------------------------------
// Y[M,Nd] = (A[M,Kd] @ W[Kd,Nd] + bias) * scale   (A fp32 or f16; out f16/f32)
// Block: 256 threads (8 waves), 128x128 tile, each wave a 32x64 subtile.
// ---------------------------------------------------------------------------
template <typename TA, bool OUT_HALF>
__global__ __launch_bounds__(256) void gemm_bias(const TA* __restrict__ A,
                                                 const float* __restrict__ W,
                                                 const float* __restrict__ bias,
                                                 void* __restrict__ out,
                                                 float scale, int M, int Nd, int Kd) {
  __shared__ __align__(16) _Float16 As[128 * 40];
  __shared__ __align__(16) _Float16 Bs[32 * 136];

  const int tid = threadIdx.x;
  const int lane = tid & 31;
  const int wv = tid >> 5;
  const int wm = (wv >> 1) * 32;   // wave M offset in tile (0,32,64,96)
  const int wn = (wv & 1) * 64;    // wave N offset in tile (0,64)
  const int blockM = blockIdx.y * 128;
  const int blockN = blockIdx.x * 128;

  v8f acc[2][4] = {};

  for (int kt = 0; kt < Kd; kt += 32) {
    // stage A tile 128x32 (convert to f16)
    for (int i = tid; i < 128 * 32; i += 256) {
      int r = i >> 5, c = i & 31;
      As[r * 40 + c] = (_Float16)A[(size_t)(blockM + r) * Kd + kt + c];
    }
    // stage W tile 32x128
    for (int i = tid; i < 32 * 128; i += 256) {
      int r = i >> 7, c = i & 127;
      Bs[r * 136 + c] = (_Float16)W[(size_t)(kt + r) * Nd + blockN + c];
    }
    __syncthreads();

    v16h a0 = frag_mk(As, wm, 40, 0);
    v16h a1 = frag_mk(As, wm + 16, 40, 0);
#pragma unroll
    for (int t = 0; t < 4; ++t) {
      v16h b = frag_kn(Bs, wn + t * 16, 136, 0);
      acc[0][t] = wmma16(a0, b, acc[0][t]);
      acc[1][t] = wmma16(a1, b, acc[1][t]);
    }
    __syncthreads();
  }

  // epilogue
#pragma unroll
  for (int ti = 0; ti < 2; ++ti)
#pragma unroll
    for (int tj = 0; tj < 4; ++tj) {
      int nc = blockN + wn + tj * 16 + (lane & 15);
      float bv = bias[nc];
      v8f a = acc[ti][tj];
#pragma unroll
      for (int r = 0; r < 8; ++r) {
        int mr = blockM + wm + ti * 16 + r + ((lane >> 4) << 3);
        float v = (a[r] + bv) * scale;
        if (OUT_HALF)
          ((_Float16*)out)[(size_t)mr * Nd + nc] = (_Float16)v;
        else
          ((float*)out)[(size_t)mr * Nd + nc] = v;
      }
    }
}

// ---------------------------------------------------------------------------
// Causal flash attention per (batch, head).  Per-head Q/K/V are contiguous
// (L, C) blocks of the flat projected buffers (faithful raw-reshape layout).
// Block: 128 threads (4 waves); wave handles 16 Q rows; 32-key chunks.
// K/V tiles double-buffered via CDNA5 async copies (software pipeline).
// Scale is already folded into the Q and K projections (as in the reference).
// ---------------------------------------------------------------------------
__global__ __launch_bounds__(128) void flash_attn(const _Float16* __restrict__ Qp,
                                                  const _Float16* __restrict__ Kp,
                                                  const _Float16* __restrict__ Vp,
                                                  const int* __restrict__ key_mask,
                                                  _Float16* __restrict__ Ctx) {
  __shared__ __align__(16) _Float16 Qs[64 * 64];
  __shared__ __align__(16) _Float16 Ks[2][32 * 64];
  __shared__ __align__(16) _Float16 Vs[2][32 * 64];
  __shared__ __align__(16) _Float16 Ps[4 * 16 * 40];

  const int tid = threadIdx.x;
  const int lane = tid & 31;
  const int w = tid >> 5;
  const int rt = blockIdx.x;                 // 64-row tile index
  const size_t headOff = ((size_t)blockIdx.z * NHEAD + blockIdx.y) * (size_t)LSEQ * CDIM;
  const _Float16* Qh = Qp + headOff;
  const _Float16* Kh = Kp + headOff;
  const _Float16* Vh = Vp + headOff;
  _Float16* Oh = Ctx + headOff;
  const int* km = key_mask + blockIdx.z * LSEQ;

  const int qrow0 = rt * 64 + w * 16;
  const int nchunk = rt * 2 + 2;             // causal: keys 0 .. rt*64+63

  // prologue: async-stage the 64x64 Q tile and chunk 0's K/V into buffer 0
  async_q_tile(Qh + (size_t)(rt * 64) * CDIM, Qs, tid);
  async_kv_tile(Kh, Vh, Ks[0], Vs[0], tid);
  asm volatile("s_wait_asynccnt 0x0" ::: "memory");
  __syncthreads();

  // this wave's Q fragments (16 x 64) from LDS
  v16h aq0 = frag_mk(Qs, w * 16, 64, 0);
  v16h aq1 = frag_mk(Qs, w * 16, 64, 32);

  float mrun[8], lrun[8];
#pragma unroll
  for (int r = 0; r < 8; ++r) { mrun[r] = -1e30f; lrun[r] = 0.0f; }
  v8f o0 = {}, o1 = {}, o2 = {}, o3 = {};

  for (int ch = 0; ch < nchunk; ++ch) {
    const int k0 = ch * 32;
    const int cur = ch & 1;

    // issue next chunk's async copies into the other buffer, then wait for
    // the current chunk (async loads complete in order: <=4 outstanding
    // means the previous 4 are done)
    if (ch + 1 < nchunk) {
      const int kn = (ch + 1) * 32;
      async_kv_tile(Kh + (size_t)kn * CDIM, Vh + (size_t)kn * CDIM,
                    Ks[cur ^ 1], Vs[cur ^ 1], tid);
      asm volatile("s_wait_asynccnt 0x4" ::: "memory");
    } else {
      asm volatile("s_wait_asynccnt 0x0" ::: "memory");
    }
    __syncthreads();                         // all waves' tile portions visible

    if (k0 <= qrow0 + 15) {                  // wave has unmasked work here
      const _Float16* Kc = Ks[cur];
      const _Float16* Vc = Vs[cur];

      // S (16 x 32) = Q(16x64) @ K^T : 4 WMMAs
      v8f s0 = {}, s1 = {};
      s0 = wmma16(aq0, frag_mk(Kc, 0, 64, 0), s0);
      s0 = wmma16(aq1, frag_mk(Kc, 0, 64, 32), s0);
      s1 = wmma16(aq0, frag_mk(Kc, 16, 64, 0), s1);
      s1 = wmma16(aq1, frag_mk(Kc, 16, 64, 32), s1);

      // causal + key mask
      const int qi = qrow0 + ((lane >> 4) << 3);
      const int kg0 = k0 + (lane & 15);
      const int kg1 = kg0 + 16;
      const int mk0 = km[kg0];
      const int mk1 = km[kg1];
#pragma unroll
      for (int r = 0; r < 8; ++r) {
        int q = qi + r;
        if (!(mk0 && kg0 <= q)) s0[r] = -1e30f;
        if (!(mk1 && kg1 <= q)) s1[r] = -1e30f;
      }

      // online softmax (per-row reductions across 16 lanes)
      float p0[8], p1[8];
#pragma unroll
      for (int r = 0; r < 8; ++r) {
        float cm = fmaxf(s0[r], s1[r]);
        cm = fmaxf(cm, __shfl_xor(cm, 1));
        cm = fmaxf(cm, __shfl_xor(cm, 2));
        cm = fmaxf(cm, __shfl_xor(cm, 4));
        cm = fmaxf(cm, __shfl_xor(cm, 8));
        float mnew = fmaxf(mrun[r], cm);
        float corr = __expf(mrun[r] - mnew);
        mrun[r] = mnew;
        lrun[r] *= corr;
        o0[r] *= corr; o1[r] *= corr; o2[r] *= corr; o3[r] *= corr;
        p0[r] = __expf(s0[r] - mnew);
        p1[r] = __expf(s1[r] - mnew);
        float rs = p0[r] + p1[r];
        rs += __shfl_xor(rs, 1);
        rs += __shfl_xor(rs, 2);
        rs += __shfl_xor(rs, 4);
        rs += __shfl_xor(rs, 8);
        lrun[r] += rs;
      }

      // D-layout -> A-layout transpose of P through per-wave LDS region
      _Float16* pw = Ps + w * 16 * 40;
      {
        int prow = (lane >> 4) << 3;
        int pc = lane & 15;
#pragma unroll
        for (int r = 0; r < 8; ++r) {
          pw[(prow + r) * 40 + pc] = (_Float16)p0[r];
          pw[(prow + r) * 40 + 16 + pc] = (_Float16)p1[r];
        }
      }
      asm volatile("s_wait_dscnt 0x0" ::: "memory");   // wave-local LDS RAW fence
      v16h ap = frag_mk(pw, 0, 40, 0);

      // O(16x64) += P(16x32) @ V(32x64) : 4 WMMAs
      o0 = wmma16(ap, frag_kn(Vc, 0, 64, 0), o0);
      o1 = wmma16(ap, frag_kn(Vc, 16, 64, 0), o1);
      o2 = wmma16(ap, frag_kn(Vc, 32, 64, 0), o2);
      o3 = wmma16(ap, frag_kn(Vc, 48, 64, 0), o3);
    }
    __syncthreads();   // nobody refills a buffer that is still being read
  }

  // normalize and write contexts (f16, faithful flat layout)
  const int nc = lane & 15;
#pragma unroll
  for (int r = 0; r < 8; ++r) {
    int mr = qrow0 + r + ((lane >> 4) << 3);
    float inv = 1.0f / lrun[r];
    Oh[(size_t)mr * CDIM + nc]      = (_Float16)(o0[r] * inv);
    Oh[(size_t)mr * CDIM + 16 + nc] = (_Float16)(o1[r] * inv);
    Oh[(size_t)mr * CDIM + 32 + nc] = (_Float16)(o2[r] * inv);
    Oh[(size_t)mr * CDIM + 48 + nc] = (_Float16)(o3[r] * inv);
  }
}

extern "C" void kernel_launch(void* const* d_in, const int* in_sizes, int n_in,
                              void* d_out, int out_size, void* d_ws, size_t ws_size,
                              hipStream_t stream) {
  const float* Q  = (const float*)d_in[0];
  const float* K  = (const float*)d_in[1];
  const float* V  = (const float*)d_in[2];
  const int* mask = (const int*)d_in[3];
  const float* Wq = (const float*)d_in[4];
  const float* bq = (const float*)d_in[5];
  const float* Wk = (const float*)d_in[6];
  const float* bk = (const float*)d_in[7];
  const float* Wv = (const float*)d_in[8];
  const float* bv = (const float*)d_in[9];
  const float* Wo = (const float*)d_in[10];
  const float* bo = (const float*)d_in[11];
  float* out = (float*)d_out;

  const int M = NB * LSEQ;                    // 8192
  const size_t NLH = (size_t)M * HDIM;        // 8M elements
  // workspace: Qp | Kp | Vp | Ctx  (f16) = 4 * 16 MB = 64 MB
  _Float16* Qp  = (_Float16*)d_ws;
  _Float16* Kp  = Qp + NLH;
  _Float16* Vp  = Kp + NLH;
  _Float16* Ctx = Vp + NLH;

  const float scale = 0.03125f;               // 1/sqrt(H), applied to Q and K

  dim3 gg(HDIM / 128, M / 128);               // (8, 64)
  gemm_bias<float, true><<<gg, 256, 0, stream>>>(Q, Wq, bq, (void*)Qp, scale, M, HDIM, HDIM);
  gemm_bias<float, true><<<gg, 256, 0, stream>>>(K, Wk, bk, (void*)Kp, scale, M, HDIM, HDIM);
  gemm_bias<float, true><<<gg, 256, 0, stream>>>(V, Wv, bv, (void*)Vp, 1.0f, M, HDIM, HDIM);

  flash_attn<<<dim3(LSEQ / 64, NHEAD, NB), 128, 0, stream>>>(Qp, Kp, Vp, mask, Ctx);

  gemm_bias<_Float16, false><<<gg, 256, 0, stream>>>(Ctx, Wo, bo, (void*)out, 1.0f, M, HDIM, HDIM);
}
